// GATConv_86852828660281
// MI455X (gfx1250) — matrix operand order repaired and checked by reading, compiled, not run on hardware
//
#include <hip/hip_runtime.h>
#include <hip/hip_bf16.h>

typedef __attribute__((ext_vector_type(16))) _Float16 v16h;
typedef __attribute__((ext_vector_type(8)))  _Float16 v8h;
typedef __attribute__((ext_vector_type(8)))  float    v8f;

#define HD    128   // H*D
#define NHEAD 4
#define DHEAD 32

// ---------------- prep kernels ----------------

__global__ void prep_xh_kernel(const float* __restrict__ x,
                               _Float16* __restrict__ xh, int total) {
    int t = blockIdx.x * blockDim.x + threadIdx.x;
    if (t < total) xh[t] = (_Float16)x[t];
}

// Wt[n][k] = (n<128 ? W_src[k][n] : W_dst[k][n-128]) as fp16, n in [0,256), k in [0,128)
__global__ void prep_wt_kernel(const float* __restrict__ Wsrc,
                               const float* __restrict__ Wdst,
                               _Float16* __restrict__ Wt) {
    int t = blockIdx.x * blockDim.x + threadIdx.x;
    if (t >= 256 * HD) return;
    int n = t >> 7;
    int k = t & 127;
    float v = (n < HD) ? Wsrc[k * HD + n] : Wdst[k * HD + (n - HD)];
    Wt[t] = (_Float16)v;
}

__global__ void init_kernel(float* __restrict__ out, float* __restrict__ smax,
                            float* __restrict__ denom, int n) {
    int t = blockIdx.x * blockDim.x + threadIdx.x;
    if (t < n * HD) out[t] = 0.0f;
    if (t < n * NHEAD) {
        smax[t]  = -__builtin_inff();
        denom[t] = 0.0f;
    }
}

// ---------------- WMMA GEMM: f[N][256] = xh @ [W_src|W_dst] + bias ----------------
// grid = (ceil(N/16)), block = 256 (8 waves). Wave w computes a 16x32 output slab:
// rows [16*bx, +16), cols [32*w, +32) as two 16x16 WMMA tiles sharing the A fragment.
// K = 128 -> 8x v_wmma_f32_16x16x32_f16 per wave, A loaded once per K-step.

__global__ __launch_bounds__(256) void gemm_wmma_kernel(
    const _Float16* __restrict__ xh,   // [N][128] row-major fp16
    const _Float16* __restrict__ Wt,   // [256][128] row-major fp16 (pre-transposed)
    const float* __restrict__ bsrc, const float* __restrict__ bdst,
    float* __restrict__ f, int n) {

    const int lane   = threadIdx.x & 31;
    const int wave   = threadIdx.x >> 5;
    const int halfLo = lane & 15;   // 0..15
    const int laneHi = lane >> 4;   // 0 or 1
    const int m0     = blockIdx.x * 16;
    const int n0     = wave * 32;   // two 16-wide column tiles: n0 and n0+16

    int row = m0 + halfLo;
    if (row >= n) row = n - 1;      // branchless clamp for tail blocks
    // A operand (16x32 fp16): lanes 0-15 hold K = kb+[0..15];
    //                          lanes 16-31 hold K = kb+[16..31].
    const _Float16* ap  = xh + (size_t)row * HD + laneHi * 16;
    // B operand (32x16 fp16): lane column = n0+halfLo; half-wave selects 16-wide K run.
    const _Float16* bp0 = Wt + (size_t)(n0 + halfLo) * HD + laneHi * 16;
    const _Float16* bp1 = bp0 + (size_t)16 * HD;

    v8f acc0 = {}, acc1 = {};
#pragma unroll
    for (int kb = 0; kb < HD; kb += 32) {
        v8h alo = *(const v8h*)(ap + kb);        // K = kb + base + [0..7]
        v8h ahi = *(const v8h*)(ap + kb + 8);    // K = kb + base + [8..15]
        v16h a;
#pragma unroll
        for (int j = 0; j < 8; ++j) { a[j] = alo[j]; a[j + 8] = ahi[j]; }
        v16h b0 = *(const v16h*)(bp0 + kb);      // 16 contiguous K, col tile 0
        v16h b1 = *(const v16h*)(bp1 + kb);      // 16 contiguous K, col tile 1
        acc0 = __builtin_amdgcn_wmma_f32_16x16x32_f16(
                   false, a, false, b0, (short)0, acc0, false, false);
        acc1 = __builtin_amdgcn_wmma_f32_16x16x32_f16(
                   false, a, false, b1, (short)0, acc1, false, false);
    }

    // C/D layout: vgpr r, lanes 0-15 -> M = r; lanes 16-31 -> M = 8+r; N = lane&15.
    const int nc0 = n0 + halfLo;
    const int nc1 = nc0 + 16;
    const float bias0 = (nc0 < HD) ? bsrc[nc0] : bdst[nc0 - HD];
    const float bias1 = (nc1 < HD) ? bsrc[nc1] : bdst[nc1 - HD];
    float* frow = f + (size_t)(m0 + laneHi * 8) * 256;

    if (m0 + 16 <= n) {
        // full tile: straight-line stores, row offset folds into the imm offset
#pragma unroll
        for (int r = 0; r < 8; ++r) {
            frow[(size_t)r * 256 + nc0] = acc0[r] + bias0;
            frow[(size_t)r * 256 + nc1] = acc1[r] + bias1;
        }
    } else {
        // tail block: guarded stores
#pragma unroll
        for (int r = 0; r < 8; ++r) {
            int m = m0 + laneHi * 8 + r;
            if (m < n) {
                frow[(size_t)r * 256 + nc0] = acc0[r] + bias0;
                frow[(size_t)r * 256 + nc1] = acc1[r] + bias1;
            }
        }
    }
}

// ---------------- edge phase ----------------

// Race-safe fp32 atomic max via sign-split integer atomics (monotone bit patterns).
__device__ inline void atomicMaxF(float* addr, float val) {
    if (val >= 0.0f) atomicMax((int*)addr, __float_as_int(val));
    else             atomicMin((unsigned int*)addr, __float_as_uint(val));
}

// One wave per edge; lane = d. score[e][h] = sum_d attn[h][d]*leaky(fs[s]+fd[d]).
__global__ __launch_bounds__(256) void edge_score_kernel(
    const int* __restrict__ src, const int* __restrict__ dst,
    const float* __restrict__ f, const float* __restrict__ attn,
    float* __restrict__ score, float* __restrict__ smax, int E) {

    int e = blockIdx.x * 8 + (threadIdx.x >> 5);
    if (e >= E) return;
    int lane = threadIdx.x & 31;
    int s = src[e], d = dst[e];
    const float* fsrow = f + (size_t)s * 256;        // fs part
    const float* fdrow = f + (size_t)d * 256 + HD;   // fd part
#pragma unroll
    for (int h = 0; h < NHEAD; ++h) {
        float v = fsrow[h * DHEAD + lane] + fdrow[h * DHEAD + lane];
        v = (v > 0.0f) ? v : 0.2f * v;               // leaky_relu, slope 0.2
        v *= attn[h * DHEAD + lane];
#pragma unroll
        for (int off = 16; off > 0; off >>= 1) v += __shfl_xor(v, off, 32);
        if (lane == 0) {
            score[(size_t)e * NHEAD + h] = v;
            atomicMaxF(&smax[d * NHEAD + h], v);
        }
    }
}

// Thread per (edge, head): ex = exp(score - smax[dst]); denom += ex; score <- ex.
__global__ void edge_exp_kernel(const int* __restrict__ dst,
                                float* __restrict__ score,
                                const float* __restrict__ smax,
                                float* __restrict__ denom, int E) {
    int t = blockIdx.x * blockDim.x + threadIdx.x;
    if (t >= E * NHEAD) return;
    int e = t >> 2, h = t & 3;
    int d = dst[e];
    float ex = __expf(score[t] - smax[d * NHEAD + h]);
    score[t] = ex;
    atomicAdd(&denom[d * NHEAD + h], ex);
}

// One wave per edge: out[dst] += (ex/denom[dst]) * fs[src].
__global__ __launch_bounds__(256) void edge_agg_kernel(
    const int* __restrict__ src, const int* __restrict__ dst,
    const float* __restrict__ f, const float* __restrict__ score,
    const float* __restrict__ denom, float* __restrict__ out, int E) {

    int e = blockIdx.x * 8 + (threadIdx.x >> 5);
    if (e >= E) return;
    int lane = threadIdx.x & 31;
    int s = src[e], d = dst[e];
    const float* fsrow = f + (size_t)s * 256;
#pragma unroll
    for (int h = 0; h < NHEAD; ++h) {
        float alpha = score[(size_t)e * NHEAD + h] / denom[d * NHEAD + h];
        atomicAdd(&out[(size_t)d * HD + h * DHEAD + lane],
                  alpha * fsrow[h * DHEAD + lane]);
    }
}

__global__ void relu_kernel(float* __restrict__ out, int total) {
    int t = blockIdx.x * blockDim.x + threadIdx.x;
    if (t < total) out[t] = fmaxf(out[t], 0.0f);
}

// ---------------- launch ----------------

extern "C" void kernel_launch(void* const* d_in, const int* in_sizes, int n_in,
                              void* d_out, int out_size, void* d_ws, size_t ws_size,
                              hipStream_t stream) {
    const float* x    = (const float*)d_in[0];
    const int*   src  = (const int*)  d_in[1];
    const int*   dst  = (const int*)  d_in[2];
    const float* Wsrc = (const float*)d_in[3];
    const float* bsrc = (const float*)d_in[4];
    const float* Wdst = (const float*)d_in[5];
    const float* bdst = (const float*)d_in[6];
    const float* attn = (const float*)d_in[7];
    float* out = (float*)d_out;

    const int N = in_sizes[0] / HD;
    const int E = in_sizes[1];

    // workspace carve-up (256B aligned)
    char* ws = (char*)d_ws;
    size_t off = 0;
    auto alignup = [](size_t o) { return (o + 255) & ~(size_t)255; };
    _Float16* xh = (_Float16*)(ws + off); off = alignup(off + (size_t)N * HD * sizeof(_Float16));
    _Float16* Wt = (_Float16*)(ws + off); off = alignup(off + (size_t)256 * HD * sizeof(_Float16));
    float* f     = (float*)(ws + off);    off = alignup(off + (size_t)N * 256 * sizeof(float));
    float* score = (float*)(ws + off);    off = alignup(off + (size_t)E * NHEAD * sizeof(float));
    float* smax  = (float*)(ws + off);    off = alignup(off + (size_t)N * NHEAD * sizeof(float));
    float* denom = (float*)(ws + off);    off = alignup(off + (size_t)N * NHEAD * sizeof(float));
    (void)ws_size; (void)n_in; (void)out_size;

    prep_xh_kernel<<<(N * HD + 255) / 256, 256, 0, stream>>>(x, xh, N * HD);
    prep_wt_kernel<<<(256 * HD + 255) / 256, 256, 0, stream>>>(Wsrc, Wdst, Wt);
    init_kernel<<<(N * HD + 255) / 256, 256, 0, stream>>>(out, smax, denom, N);

    gemm_wmma_kernel<<<(N + 15) / 16, 256, 0, stream>>>(xh, Wt, bsrc, bdst, f, N);

    edge_score_kernel<<<(E + 7) / 8, 256, 0, stream>>>(src, dst, f, attn, score, smax, E);
    edge_exp_kernel<<<(E * NHEAD + 255) / 256, 256, 0, stream>>>(dst, score, smax, denom, E);
    edge_agg_kernel<<<(E + 7) / 8, 256, 0, stream>>>(src, dst, f, score, denom, out, E);
    relu_kernel<<<(N * HD + 255) / 256, 256, 0, stream>>>(out, N * HD);
}
